// DenseQTripletLoss_69672959475806
// MI455X (gfx1250) — matrix operand order repaired and checked by reading, not compiled, not run on hardware
//
#include <hip/hip_runtime.h>

// ---------------- problem constants ----------------
#define Bn   16
#define Cc   256
#define Hh   320
#define Ww   320
#define GS   8
#define HC   40
#define WC   40
#define Nn   1600          // HC*WC cells
#define NT   100           // Nn/16 tiles
#define CPAD 264           // 256 + 8 bf16 pad -> conflict-free ds_load_b128

typedef __bf16 bf16;
typedef __attribute__((ext_vector_type(4)))  bf16  v4bf;
typedef __attribute__((ext_vector_type(8)))  bf16  v8bf;
typedef __attribute__((ext_vector_type(16))) bf16  v16bf;
typedef __attribute__((ext_vector_type(8)))  float v8f;

// Stage a 4-channel x 4-row block: global float4 loads (contiguous over n),
// register transpose, packed 4x-bf16 hi/lo stores (ds_store_b64).
__device__ __forceinline__
void stage_block(const float* __restrict__ gcol0,   // &d[b][c0][n_base + r0]
                 bf16* __restrict__ hi, bf16* __restrict__ lo,
                 int c0, int r0) {
  float4 q[4];
  #pragma unroll
  for (int cc = 0; cc < 4; ++cc)
    q[cc] = *(const float4*)(gcol0 + (size_t)cc * Nn);
  float va[4][4] = { { q[0].x, q[0].y, q[0].z, q[0].w },
                     { q[1].x, q[1].y, q[1].z, q[1].w },
                     { q[2].x, q[2].y, q[2].z, q[2].w },
                     { q[3].x, q[3].y, q[3].z, q[3].w } };
  #pragma unroll
  for (int j = 0; j < 4; ++j) {
    v4bf ph, pl;
    #pragma unroll
    for (int cc = 0; cc < 4; ++cc) {
      float x = va[cc][j];
      bf16 h = (bf16)x;
      ph[cc] = h;
      pl[cc] = (bf16)(x - (float)h);
    }
    *(v4bf*)&hi[(r0 + j) * CPAD + c0] = ph;
    *(v4bf*)&lo[(r0 + j) * CPAD + c0] = pl;
  }
}

// ---------------- kernel 1: warp coords + wv_match, zero accumulators -------
__global__ void k_coords(const float* __restrict__ homo,
                         float* __restrict__ wy, float* __restrict__ wx,
                         float* __restrict__ wv, float* __restrict__ acc) {
  int idx = blockIdx.x * blockDim.x + threadIdx.x;
  if (idx == 0) { acc[0] = 0.f; acc[1] = 0.f; }
  if (idx >= Bn * Nn) return;
  int b = idx / Nn, n = idx % Nn;
  float x = (float)((n % WC) * GS + GS / 2);
  float y = (float)((n / WC) * GS + GS / 2);
  const float* h = homo + b * 9;
  float px = h[0] * x + h[1] * y + h[2];
  float py = h[3] * x + h[4] * y + h[5];
  float pz = h[6] * x + h[7] * y + h[8];
  float wxi = px / pz, wyi = py / pz;
  wy[idx] = wyi; wx[idx] = wxi;
  wv[idx] = (wyi > 0.f && wyi < (float)(Hh - 1) &&
             wxi > 0.f && wxi < (float)(Ww - 1)) ? 1.f : 0.f;
}

// ---------------- kernel 2: 8x8 visibility product per cell -----------------
__global__ void k_vis(const float* __restrict__ mask, float* __restrict__ vis) {
  int idx = blockIdx.x * blockDim.x + threadIdx.x;
  if (idx >= Bn * Nn) return;
  int b = idx / Nn, n = idx % Nn;
  int cy = n / WC, cx = n % WC;
  const float* mp = mask + ((size_t)b * Hh + (size_t)cy * GS) * Ww + (size_t)cx * GS;
  float p = 1.f;
  #pragma unroll
  for (int i = 0; i < GS; ++i) {
    float4 a = *(const float4*)(mp + i * Ww);
    float4 c = *(const float4*)(mp + i * Ww + 4);
    p *= a.x * a.y * a.z * a.w * c.x * c.y * c.z * c.w;
  }
  vis[idx] = p;
}

// ---------------- kernel 3: pos_sim via bilinear sample (1 wave / point) ----
__global__ void k_pos(const float* __restrict__ desc1, const float* __restrict__ desc2,
                      const float* __restrict__ wy, const float* __restrict__ wx,
                      float* __restrict__ pos) {
  int lane = threadIdx.x & 31;
  int wave = threadIdx.x >> 5;
  int p = blockIdx.x * 8 + wave;   // blockDim.x == 256 -> 8 waves
  if (p >= Bn * Nn) return;
  int b = p / Nn, n = p % Nn;
  float cy = wy[p] * (1.f / GS) - 0.5f;
  float cx = wx[p] * (1.f / GS) - 0.5f;
  float y0f = floorf(cy), x0f = floorf(cx);
  float fy = cy - y0f, fx = cx - x0f;
  int y0 = (int)y0f, x0 = (int)x0f;
  float w4[4] = { (1.f - fy) * (1.f - fx), (1.f - fy) * fx,
                  fy * (1.f - fx),         fy * fx };
  int ys[4] = { y0, y0, y0 + 1, y0 + 1 };
  int xs[4] = { x0, x0 + 1, x0, x0 + 1 };
  float cw[4]; int coff[4];
  #pragma unroll
  for (int k = 0; k < 4; ++k) {
    int yi = ys[k], xi = xs[k];
    bool v = (yi >= 0 && yi < HC && xi >= 0 && xi < WC);
    int yc = yi < 0 ? 0 : (yi > HC - 1 ? HC - 1 : yi);
    int xc = xi < 0 ? 0 : (xi > WC - 1 ? WC - 1 : xi);
    cw[k] = v ? w4[k] : 0.f;
    coff[k] = yc * WC + xc;
  }
  const float* d2b = desc2 + (size_t)b * Cc * Nn;
  const float* d1p = desc1 + (size_t)b * Cc * Nn + n;
  float dot = 0.f, nrm = 0.f;
  #pragma unroll
  for (int r = 0; r < 8; ++r) {
    int c = r * 32 + lane;
    size_t co = (size_t)c * Nn;
    float wd = cw[0] * d2b[co + coff[0]] + cw[1] * d2b[co + coff[1]] +
               cw[2] * d2b[co + coff[2]] + cw[3] * d2b[co + coff[3]];
    dot += d1p[co] * wd;
    nrm += wd * wd;
  }
  #pragma unroll
  for (int m = 16; m >= 1; m >>= 1) {
    dot += __shfl_xor(dot, m, 32);
    nrm += __shfl_xor(nrm, m, 32);
  }
  if (lane == 0) pos[p] = 2.f - 2.f * dot / fmaxf(sqrtf(nrm), 1e-12f);
}

// ---------------- kernel 4: fused WMMA GEMM + penalties + row-min -----------
// grid = Bn*NT workgroups of 64 threads (2 waves); wave w owns m-tile 2*it+w.
__global__ __launch_bounds__(64)
void k_negmin(const float* __restrict__ desc1, const float* __restrict__ desc2,
              const float* __restrict__ wyb, const float* __restrict__ wxb,
              const float* __restrict__ vis, float* __restrict__ neg) {
  __shared__ bf16 sa_hi[16 * CPAD], sa_lo[16 * CPAD];
  __shared__ bf16 sb_hi[2][16 * CPAD], sb_lo[2][16 * CPAD];
  __shared__ float s_wy[16], s_wx[16], s_rmin[2][16];

  int tid  = threadIdx.x;
  int wave = tid >> 5, lane = tid & 31;
  int b  = blockIdx.x / NT;
  int nt = blockIdx.x % NT;
  int n0 = nt * 16;
  const float* d1b = desc1 + (size_t)b * Cc * Nn;
  const float* d2b = desc2 + (size_t)b * Cc * Nn;

  // stage A tile (16 n-rows x 256 c): 256 blocks of (4c x 4n), 4 per thread
  #pragma unroll
  for (int i = 0; i < 4; ++i) {
    int idx = i * 64 + tid;            // [0, 256)
    int c0 = (idx >> 2) * 4;
    int r0 = (idx & 3) * 4;
    stage_block(d1b + (size_t)c0 * Nn + n0 + r0, sa_hi, sa_lo, c0, r0);
  }
  if (tid < 16) {
    s_wy[tid] = wyb[b * Nn + n0 + tid];
    s_wx[tid] = wxb[b * Nn + n0 + tid];
  }

  int r  = lane & 15;
  int lh = lane >> 4;                 // lane-half (0/1)
  const bf16* arow_h = sa_hi + r * CPAD;
  const bf16* arow_l = sa_lo + r * CPAD;
  const bf16* brow_h = sb_hi[wave] + r * CPAD;
  const bf16* brow_l = sb_lo[wave] + r * CPAD;

  float rmin[8];
  #pragma unroll
  for (int v = 0; v < 8; ++v) rmin[v] = 3.0e38f;

  for (int it = 0; it < NT / 2; ++it) {
    __syncthreads();
    int mbase = it * 32;
    // stage both waves' B tiles: 512 blocks of (4c x 4n), 8 per thread
    #pragma unroll
    for (int i = 0; i < 8; ++i) {
      int idx = i * 64 + tid;          // [0, 512)
      int t  = idx >> 8;               // which tile (0/1)
      int j  = idx & 255;
      int c0 = (j >> 2) * 4;
      int r0 = (j & 3) * 4;
      stage_block(d2b + (size_t)c0 * Nn + mbase + t * 16 + r0,
                  sb_hi[t], sb_lo[t], c0, r0);
    }
    __syncthreads();

    v8f acc = {};
    #pragma unroll
    for (int kb = 0; kb < Cc; kb += 32) {
      int ka  = kb + lh * 16;         // A: lane<16 -> K{0..15} contiguous
      int kbb = kb + lh * 16;         // B: lane<16 -> K{0..15} contiguous
      union U { v16bf v; v8bf h[2]; };
      U ah, al, bh, bl;
      ah.h[0] = *(const v8bf*)(arow_h + ka);      ah.h[1] = *(const v8bf*)(arow_h + ka + 8);
      al.h[0] = *(const v8bf*)(arow_l + ka);      al.h[1] = *(const v8bf*)(arow_l + ka + 8);
      bh.h[0] = *(const v8bf*)(brow_h + kbb);     bh.h[1] = *(const v8bf*)(brow_h + kbb + 8);
      bl.h[0] = *(const v8bf*)(brow_l + kbb);     bl.h[1] = *(const v8bf*)(brow_l + kbb + 8);
      // bf16x3 error-compensated fp32 dot
      acc = __builtin_amdgcn_wmma_f32_16x16x32_bf16(false, ah.v, false, bh.v, (short)0, acc, false, false);
      acc = __builtin_amdgcn_wmma_f32_16x16x32_bf16(false, ah.v, false, bl.v, (short)0, acc, false, false);
      acc = __builtin_amdgcn_wmma_f32_16x16x32_bf16(false, al.v, false, bh.v, (short)0, acc, false, false);
    }

    // penalties + running per-row min. C layout: VGPR v -> row v+8*lh, col r.
    int m  = mbase + wave * 16 + r;
    int my = m / WC, mx = m % WC;
    float gy = (float)(my * GS + GS / 2);
    float gx = (float)(mx * GS + GS / 2);
    float penv = 5.f * (1.f - vis[b * Nn + m]);
    #pragma unroll
    for (int v = 0; v < 8; ++v) {
      int nl = lh * 8 + v;
      float dy = s_wy[nl] - gy, dx = s_wx[nl] - gx;
      float dist = sqrtf(fmaxf(dy * dy + dx * dx, 1e-12f));
      float s = 2.f - 2.f * acc[v] + ((dist <= (float)GS) ? 5.f : 0.f) + penv;
      rmin[v] = fminf(rmin[v], s);
    }
  }

  // min across the 16 lanes of each half, then across waves via LDS
  #pragma unroll
  for (int v = 0; v < 8; ++v) {
    #pragma unroll
    for (int msk = 1; msk < 16; msk <<= 1)
      rmin[v] = fminf(rmin[v], __shfl_xor(rmin[v], msk, 32));
  }
  if (r == 0) {
    #pragma unroll
    for (int v = 0; v < 8; ++v) s_rmin[wave][lh * 8 + v] = rmin[v];
  }
  __syncthreads();
  if (tid < 16) neg[b * Nn + n0 + tid] = fminf(s_rmin[0][tid], s_rmin[1][tid]);
}

// ---------------- kernel 5: hinge loss partial sums -------------------------
__global__ void k_loss(const float* __restrict__ pos, const float* __restrict__ neg,
                       const float* __restrict__ wv, float* __restrict__ acc) {
  __shared__ float sl[256], sw[256];
  int idx = blockIdx.x * 256 + threadIdx.x;
  float l = 0.f, w = 0.f;
  if (idx < Bn * Nn) {
    w = wv[idx];
    float t = fmaxf(pos[idx] - neg[idx] + 1.0f, 0.f);
    l = t * t * w;
  }
  sl[threadIdx.x] = l; sw[threadIdx.x] = w;
  __syncthreads();
  for (int s = 128; s > 0; s >>= 1) {
    if (threadIdx.x < s) {
      sl[threadIdx.x] += sl[threadIdx.x + s];
      sw[threadIdx.x] += sw[threadIdx.x + s];
    }
    __syncthreads();
  }
  if (threadIdx.x == 0) { atomicAdd(&acc[0], sl[0]); atomicAdd(&acc[1], sw[0]); }
}

__global__ void k_final(const float* __restrict__ acc, float* __restrict__ out) {
  out[0] = acc[0] / acc[1];   // LOSS_LAMBDA == 1
}

// ---------------- launcher --------------------------------------------------
extern "C" void kernel_launch(void* const* d_in, const int* in_sizes, int n_in,
                              void* d_out, int out_size, void* d_ws, size_t ws_size,
                              hipStream_t stream) {
  const float* desc1 = (const float*)d_in[0];
  const float* desc2 = (const float*)d_in[1];
  const float* homo  = (const float*)d_in[2];
  const float* vmask = (const float*)d_in[3];
  // d_in[4] (score2) only contributes its shape; unused numerically.

  float* ws = (float*)d_ws;
  const int BN = Bn * Nn;           // 25600
  float* wy  = ws + 0 * BN;
  float* wx  = ws + 1 * BN;
  float* wv  = ws + 2 * BN;
  float* vis = ws + 3 * BN;
  float* pos = ws + 4 * BN;
  float* neg = ws + 5 * BN;
  float* acc = ws + 6 * BN;         // 2 floats

  k_coords<<<BN / 256, 256, 0, stream>>>(homo, wy, wx, wv, acc);
  k_vis   <<<BN / 256, 256, 0, stream>>>(vmask, vis);
  k_pos   <<<BN / 8,   256, 0, stream>>>(desc1, desc2, wy, wx, pos);
  k_negmin<<<Bn * NT,   64, 0, stream>>>(desc1, desc2, wy, wx, vis, neg);
  k_loss  <<<BN / 256, 256, 0, stream>>>(pos, neg, wv, acc);
  k_final <<<1, 1, 0, stream>>>(acc, (float*)d_out);
}
